// RITS_66675072303366
// MI455X (gfx1250) — matrix operand order repaired and checked by reading, not compiled, not run on hardware
//
#include <hip/hip_runtime.h>

typedef __attribute__((ext_vector_type(16))) _Float16 v16h;
typedef __attribute__((ext_vector_type(8)))  float    v8f;

#define B_TOT    1024
#define T_LEN    512
#define D_IN     2
#define H_DIM    128
#define G4       512      // 4*H
#define ROWS     16       // batch rows per workgroup
#define NTHREADS 512      // 16 waves

__device__ __forceinline__ float sigmoidf_(float x) {
    return 1.0f / (1.0f + __expf(-x));      // branch-free: v_exp + v_rcp
}
__device__ __forceinline__ float tanhf_(float x) {
    // branch-free tanh via exp; clamp keeps exp finite (exp(30) ~ 1e13)
    float xc = fminf(fmaxf(x, -15.f), 15.f);
    float e  = __expf(2.f * xc);
    return (e - 1.f) / (e + 1.f);
}

// -------- pass 1: per-timestep inverse denominator (batch-wide mask sum) ----
__global__ void rits_denom_kernel(const float* __restrict__ masks,
                                  float* __restrict__ inv_denom) {
    __shared__ float red[256];
    const int t = blockIdx.x;
    const int tid = threadIdx.x;
    float s = 0.f;
    #pragma unroll
    for (int k = 0; k < 2048; k += 256) {
        int idx = tid + k;              // 0..2047 over (b, c)
        int b = idx >> 1, c = idx & 1;
        s += masks[b * (T_LEN * D_IN) + t * D_IN + c];
    }
    red[tid] = s;
    __syncthreads();
    for (int off = 128; off > 0; off >>= 1) {
        if (tid < off) red[tid] += red[tid + off];
        __syncthreads();
    }
    if (tid == 0) inv_denom[t] = 1.0f / (red[0] + 1e-6f);
}

// -------- pass 2: the full recurrent scan, 16 batch rows per workgroup ------
__global__ __launch_bounds__(NTHREADS)
void rits_main_kernel(const float* __restrict__ values,
                      const float* __restrict__ masks,
                      const float* __restrict__ deltas,
                      const float* __restrict__ tdh_W,  const float* __restrict__ tdh_b,
                      const float* __restrict__ tdx_W,  const float* __restrict__ tdx_b,
                      const float* __restrict__ hist_W, const float* __restrict__ hist_b,
                      const float* __restrict__ fr_W,   const float* __restrict__ fr_b,
                      const float* __restrict__ wc_W,   const float* __restrict__ wc_b,
                      const float* __restrict__ lstm_Wk,
                      const float* __restrict__ lstm_Wr,
                      const float* __restrict__ lstm_b,
                      const float* __restrict__ inv_denom,
                      float* __restrict__ out_imp,
                      float* __restrict__ out_loss)
{
    __shared__ __align__(16) float    sh_Wk[4 * G4];        // 8 KB
    __shared__ __align__(16) float    sh_bias[G4];          // 2 KB
    __shared__ __align__(16) float    sh_tdhW[H_DIM * 2];   // 1 KB
    __shared__ __align__(16) float    sh_tdhb[H_DIM];       // 0.5 KB
    __shared__ __align__(16) float    sh_histW[H_DIM * 2];  // 1 KB
    __shared__ __align__(16) float    sh_invden[T_LEN];     // 2 KB
    __shared__ __align__(16) float    sh_h[ROWS * H_DIM];   // 8 KB  (LSTM h, f32)
    __shared__ __align__(16) _Float16 sh_hA[ROWS * H_DIM];  // 4 KB  (decayed h, WMMA A feed)
    __shared__ __align__(16) float    sh_gates[ROWS * G4];  // 32 KB
    __shared__ __align__(16) float    sh_part[32 * 16];     // 2 KB  (x_hat tree reduction)
    __shared__ float sh_x[32], sh_m[32], sh_d[32], sh_cc[32];
    __shared__ float sh_sw[24];  // hist_b[2] fr_W[4] fr_b[2] tdx_W[4] tdx_b[2] wc_W[8] wc_b[2]

    const int tid    = threadIdx.x;
    const int lane   = tid & 31;
    const int wv     = tid >> 5;          // wave 0..15
    const int colA   = lane & 15;
    const int hiflag = lane >> 4;         // 0 or 1 (half-wave)
    const int row0   = blockIdx.x * ROWS;

    // ---- one-time LDS fills ----
    for (int i = tid; i < 4 * G4; i += NTHREADS) sh_Wk[i]   = lstm_Wk[i];
    for (int i = tid; i < G4;     i += NTHREADS) sh_bias[i] = lstm_b[i];
    for (int i = tid; i < H_DIM * 2; i += NTHREADS) sh_tdhW[i]  = tdh_W[i];
    for (int i = tid; i < H_DIM;     i += NTHREADS) sh_tdhb[i]  = tdh_b[i];
    for (int i = tid; i < H_DIM * 2; i += NTHREADS) sh_histW[i] = hist_W[i];
    for (int i = tid; i < T_LEN;     i += NTHREADS) sh_invden[i] = inv_denom[i];
    for (int i = tid; i < ROWS * H_DIM; i += NTHREADS) sh_h[i] = 0.f;
    if (tid < 24) {
        float v;
        if      (tid < 2)  v = hist_b[tid];
        else if (tid < 6)  v = fr_W[tid - 2];
        else if (tid < 8)  v = fr_b[tid - 6];
        else if (tid < 12) v = tdx_W[tid - 8];
        else if (tid < 14) v = tdx_b[tid - 12];
        else if (tid < 22) v = wc_W[tid - 14];
        else               v = wc_b[tid - 22];
        sh_sw[tid] = v;
    }

    // ---- one-time: build register-resident B fragments of lstm_Wr (f16) ----
    // B layout (16x16x32 f16): lanes 0-15 col=lane, K=0..7,16..23; lanes 16-31 col=lane-16, K=8..15,24..31
    v16h bfrag[2][4];
    #pragma unroll
    for (int tg = 0; tg < 2; ++tg) {
        const int g   = 2 * wv + tg;        // gate tile 0..31
        const int col = g * 16 + colA;
        #pragma unroll
        for (int kc = 0; kc < 4; ++kc) {
            const int kb = kc * 32 + hiflag * 8;
            v16h f;
            #pragma unroll
            for (int v = 0; v < 8; ++v) {
                f[2 * v]     = (_Float16)lstm_Wr[(kb + 2 * v     + (v >= 4 ? 8 : 0)) * G4 + col];
                f[2 * v + 1] = (_Float16)lstm_Wr[(kb + 2 * v + 1 + (v >= 4 ? 8 : 0)) * G4 + col];
            }
            bfrag[tg][kc] = f;
        }
    }

    float c_reg[4] = {0.f, 0.f, 0.f, 0.f};  // cell state, fixed thread->element map
    float loss = 0.f;                        // per-row loss (threads 0..15)

    // prefetch registers for step-t inputs (threads 0..31)
    float rx = 0.f, rm = 0.f, rd = 0.f;
    if (tid < 32) {
        int r = tid >> 1, c = tid & 1;
        int off = (row0 + r) * (T_LEN * D_IN) + c;   // t = 0
        rx = values[off];
        rm = masks[off];
        rd = deltas[off];
    }

    __syncthreads();

    for (int t = 0; t < T_LEN; ++t) {
        // stage 0: publish prefetched x, m, d for this step
        if (tid < 32) {
            sh_x[tid] = rx;
            sh_m[tid] = rm;
            sh_d[tid] = rd;
        }
        __syncthreads();

        // issue next step's global loads now; they complete during stages 1-5
        if (tid < 32 && t + 1 < T_LEN) {
            int r = tid >> 1, c = tid & 1;
            int off = (row0 + r) * (T_LEN * D_IN) + (t + 1) * D_IN + c;
            rx = values[off];
            rm = masks[off];
            rd = deltas[off];
        }

        // stage 1: temporal decay of h -> f16 A operand
        #pragma unroll
        for (int k = 0; k < 4; ++k) {
            int idx = tid + NTHREADS * k;           // 0..2047
            int r = idx >> 7, j = idx & 127;
            float a = sh_d[2 * r] * sh_tdhW[2 * j] + sh_d[2 * r + 1] * sh_tdhW[2 * j + 1]
                      + sh_tdhb[j];
            float gam = __expf(-fmaxf(a, 0.f));
            sh_hA[idx] = (_Float16)(sh_h[idx] * gam);
        }
        __syncthreads();

        // stage 2: x_hat = h @ hist_W, partial sums (deterministic tree)
        {
            int pair = tid >> 4, p = tid & 15;      // pair = (r, c)
            int r = pair >> 1, c = pair & 1;
            float s = 0.f;
            #pragma unroll
            for (int q = 0; q < 8; ++q) {
                int j = p * 8 + q;
                s += (float)sh_hA[r * H_DIM + j] * sh_histW[2 * j + c];
            }
            sh_part[pair * 16 + p] = s;
        }
        __syncthreads();

        // stage 3: finish x_hat reduction + per-row scalar chain + loss + imputation
        if (tid < 16) {
            const int r = tid;
            float xh0 = sh_sw[0], xh1 = sh_sw[1];
            #pragma unroll
            for (int p = 0; p < 16; ++p) {
                xh0 += sh_part[(2 * r)     * 16 + p];
                xh1 += sh_part[(2 * r + 1) * 16 + p];
            }
            float x0 = sh_x[2 * r], x1 = sh_x[2 * r + 1];
            float m0 = sh_m[2 * r], m1 = sh_m[2 * r + 1];
            float d0 = sh_d[2 * r], d1 = sh_d[2 * r + 1];
            float inv = sh_invden[t];
            float w0 = m0 * inv;            // ll[0] = 1.0
            float w1 = 0.5f * m1 * inv;     // ll[1] = 0.5

            float e0 = x0 - xh0, e1 = x1 - xh1;
            loss += (fabsf(e0) + e0 * e0) * w0 + (fabsf(e1) + e1 * e1) * w1;

            float xc0 = m0 * x0 + (1.f - m0) * xh0;
            float xc1 = m1 * x1 + (1.f - m1) * xh1;
            // z = x_c @ fr_W.T + fr_b  (fr_W in sh_sw[2..5], fr_b in [6..7])
            float z0 = xc0 * sh_sw[2] + xc1 * sh_sw[3] + sh_sw[6];
            float z1 = xc0 * sh_sw[4] + xc1 * sh_sw[5] + sh_sw[7];
            e0 = x0 - z0; e1 = x1 - z1;
            loss += (fabsf(e0) + e0 * e0) * w0 + (fabsf(e1) + e1 * e1) * w1;

            float gx0 = __expf(-fmaxf(d0 * sh_sw[8]  + d1 * sh_sw[9]  + sh_sw[12], 0.f));
            float gx1 = __expf(-fmaxf(d0 * sh_sw[10] + d1 * sh_sw[11] + sh_sw[13], 0.f));
            // beta = [gx, m] @ wc_W + wc_b  (wc_W in sh_sw[14..21], wc_b [22..23])
            float b0 = gx0 * sh_sw[14] + gx1 * sh_sw[16] + m0 * sh_sw[18] + m1 * sh_sw[20] + sh_sw[22];
            float b1 = gx0 * sh_sw[15] + gx1 * sh_sw[17] + m0 * sh_sw[19] + m1 * sh_sw[21] + sh_sw[23];
            float ch0 = b0 * z0 + (1.f - b0) * xh0;
            float ch1 = b1 * z1 + (1.f - b1) * xh1;
            e0 = x0 - ch0; e1 = x1 - ch1;
            loss += (fabsf(e0) + e0 * e0) * w0 + (fabsf(e1) + e1 * e1) * w1;

            float cc0 = m0 * x0 + (1.f - m0) * ch0;
            float cc1 = m1 * x1 + (1.f - m1) * ch1;
            sh_cc[2 * r] = cc0; sh_cc[2 * r + 1] = cc1;
            int o = (row0 + r) * (T_LEN * D_IN) + t * D_IN;
            out_imp[o] = cc0; out_imp[o + 1] = cc1;
        }
        __syncthreads();

        // stage 4: gates = [c_c, m] @ Wk + bias + h @ Wr  (WMMA, 8 per wave)
        {
            // A fragments of decayed h (16-bit A layout from ISA table)
            v16h afrag[4];
            const _Float16* hrow = &sh_hA[(lane & 15) * H_DIM];
            const int koff = hiflag * 8;
            #pragma unroll
            for (int kc = 0; kc < 4; ++kc) {
                union { v16h v; float4 f[2]; } u;
                u.f[0] = *(const float4*)(hrow + kc * 32 + koff);
                u.f[1] = *(const float4*)(hrow + kc * 32 + 16 + koff);
                afrag[kc] = u.v;
            }
            #pragma unroll
            for (int tg = 0; tg < 2; ++tg) {
                const int g   = 2 * wv + tg;
                const int col = g * 16 + colA;
                const float wk0 = sh_Wk[col],            wk1 = sh_Wk[G4 + col];
                const float wk2 = sh_Wk[2 * G4 + col],   wk3 = sh_Wk[3 * G4 + col];
                const float bse = sh_bias[col];
                v8f acc;
                #pragma unroll
                for (int q = 0; q < 8; ++q) {
                    int row = q + hiflag * 8;   // C layout: VGPR q -> M=q / q+8
                    acc[q] = bse + sh_cc[2 * row] * wk0 + sh_cc[2 * row + 1] * wk1
                                 + sh_m[2 * row]  * wk2 + sh_m[2 * row + 1]  * wk3;
                }
                #pragma unroll
                for (int kc = 0; kc < 4; ++kc) {
                    acc = __builtin_amdgcn_wmma_f32_16x16x32_f16(
                        false, afrag[kc], false, bfrag[tg][kc],
                        (short)0, acc, false, false);
                }
                #pragma unroll
                for (int q = 0; q < 8; ++q) {
                    int row = q + hiflag * 8;
                    sh_gates[row * G4 + col] = acc[q];
                }
            }
        }
        __syncthreads();

        // stage 5: LSTM elementwise update (c in registers, h back to LDS)
        #pragma unroll
        for (int k = 0; k < 4; ++k) {
            int idx = tid + NTHREADS * k;
            int r = idx >> 7, j = idx & 127;
            float gi = sh_gates[r * G4 + j];
            float gf = sh_gates[r * G4 + H_DIM + j];
            float gg = sh_gates[r * G4 + 2 * H_DIM + j];
            float go = sh_gates[r * G4 + 3 * H_DIM + j];
            float c = sigmoidf_(gf) * c_reg[k] + sigmoidf_(gi) * tanhf_(gg);
            c_reg[k] = c;
            sh_h[idx] = sigmoidf_(go) * tanhf_(c);
        }
        // no trailing barrier: next iteration's stage-0 barrier orders sh_h
    }

    if (tid < 16) out_loss[row0 + tid] = loss;
}

extern "C" void kernel_launch(void* const* d_in, const int* in_sizes, int n_in,
                              void* d_out, int out_size, void* d_ws, size_t ws_size,
                              hipStream_t stream) {
    const float* values  = (const float*)d_in[0];
    const float* masks   = (const float*)d_in[1];
    const float* deltas  = (const float*)d_in[2];
    const float* tdh_W   = (const float*)d_in[3];
    const float* tdh_b   = (const float*)d_in[4];
    const float* tdx_W   = (const float*)d_in[5];
    const float* tdx_b   = (const float*)d_in[6];
    const float* hist_W  = (const float*)d_in[7];
    const float* hist_b  = (const float*)d_in[8];
    const float* fr_W    = (const float*)d_in[9];
    const float* fr_b    = (const float*)d_in[10];
    const float* wc_W    = (const float*)d_in[11];
    const float* wc_b    = (const float*)d_in[12];
    const float* lstm_Wk = (const float*)d_in[13];
    const float* lstm_Wr = (const float*)d_in[14];
    const float* lstm_b  = (const float*)d_in[15];

    float* inv_denom = (float*)d_ws;                            // T_LEN floats
    float* out_imp   = (float*)d_out;                           // B*T*D
    float* out_loss  = out_imp + (size_t)B_TOT * T_LEN * D_IN;  // B

    rits_denom_kernel<<<T_LEN, 256, 0, stream>>>(masks, inv_denom);
    rits_main_kernel<<<B_TOT / ROWS, NTHREADS, 0, stream>>>(
        values, masks, deltas, tdh_W, tdh_b, tdx_W, tdx_b,
        hist_W, hist_b, fr_W, fr_b, wc_W, wc_b,
        lstm_Wk, lstm_Wr, lstm_b, inv_denom, out_imp, out_loss);
}
